// Morpher_22471268893244
// MI455X (gfx1250) — matrix operand order repaired and checked by reading, not compiled, not run on hardware
//
#include <hip/hip_runtime.h>
#include <hip/hip_bf16.h>
#include <math.h>

typedef __attribute__((ext_vector_type(16))) _Float16 v16h;
typedef __attribute__((ext_vector_type(8)))  _Float16 v8h;
typedef __attribute__((ext_vector_type(8)))  float    v8f;

#define N_STREAMS 4
#define KSLOTS    3
#define S_SLOTS   7
#define DHEAD     128
#define DMODEL    896
#define HEADS     12          // N_STREAMS * KSLOTS
#define IODIM     768
#define MIXHID    3584
#define BATCH     4
#define SEQ       1024
#define BT        (BATCH*SEQ)        // 4096
#define NBT       (N_STREAMS*BT)     // 16384
#define LN_EPS    1e-5f
#define CHUNK     128                // f16 elements per staged k-chunk

// ---------------- CDNA5 async copy (global -> LDS), ASYNCcnt-tracked -------
static __device__ __forceinline__ void async_copy_b128(unsigned lds_off, const void* gptr) {
  asm volatile("global_load_async_to_lds_b128 %0, %1, off"
               :: "v"(lds_off), "v"(gptr) : "memory");
}
static __device__ __forceinline__ void wait_async0() {
  asm volatile("s_wait_asynccnt 0x0" ::: "memory");
}

// ---------------- WMMA fragment loaders (CDNA5 16-bit layouts, wave32) -----
// A (16x32 f16): lane L holds row m=L&15; half=L>>4; elems 0..7 -> k=kb+half*8+i,
// elems 8..15 -> k=kb+half*8+16+i  (two contiguous 16B chunks)
static __device__ __forceinline__ v16h load_a_frag(const _Float16* __restrict__ A,
                                                   int lda, int m0, int kb) {
  int lane = threadIdx.x & 31;
  const _Float16* p = A + (size_t)(m0 + (lane & 15)) * lda + kb + ((lane >> 4) << 3);
  v8h lo = *(const v8h*)(p);
  v8h hi = *(const v8h*)(p + 16);
  v16h r;
#pragma unroll
  for (int i = 0; i < 8; ++i) { r[i] = lo[i]; r[i + 8] = hi[i]; }
  return r;
}

// A fragment out of an LDS-staged 16 x CHUNK strip (row stride = CHUNK)
static __device__ __forceinline__ v16h load_a_frag_lds(const _Float16* __restrict__ Ab,
                                                       int kb) {
  int lane = threadIdx.x & 31;
  const _Float16* p = Ab + (lane & 15) * CHUNK + kb + (((lane >> 4) & 1) << 3);
  v8h lo = *(const v8h*)(p);
  v8h hi = *(const v8h*)(p + 16);
  v16h r;
#pragma unroll
  for (int i = 0; i < 8; ++i) { r[i] = lo[i]; r[i + 8] = hi[i]; }
  return r;
}

// B (32x16 f16) built from row-major weights W[n][k] (B[k][n] = W[n0+n][k]):
// lane L holds col n=L&15; elems 0..7 -> k=kb+half*8+i, elems 8..15 -> k=kb+half*8+16+i
static __device__ __forceinline__ v16h load_b_frag(const _Float16* __restrict__ W,
                                                   int ldw, int n0, int kb) {
  int lane = threadIdx.x & 31;
  const _Float16* p = W + (size_t)(n0 + (lane & 15)) * ldw + kb + ((lane >> 4) << 3);
  v8h lo = *(const v8h*)(p);
  v8h hi = *(const v8h*)(p + 16);
  v16h r;
#pragma unroll
  for (int i = 0; i < 8; ++i) { r[i] = lo[i]; r[i + 8] = hi[i]; }
  return r;
}

// Stage one 16 x CHUNK f16 A-chunk into LDS cooperatively (64 threads,
// 4 async b128 per thread: 256 x 16B = 16*128*2 bytes).
static __device__ __forceinline__ void stage_a_chunk(const _Float16* __restrict__ A,
                                                     int lda, int m0, int kc,
                                                     _Float16* dstLds) {
  int tid = threadIdx.x & 63;
#pragma unroll
  for (int j = 0; j < 4; ++j) {
    int e = tid + 64 * j;          // 0..255 -> one 16B (8-half) packet each
    int row = e >> 4;              // 16 packets per row
    int col8 = e & 15;
    unsigned loff = (unsigned)(uintptr_t)(dstLds + row * CHUNK + col8 * 8);
    const void* g = A + (size_t)(m0 + row) * lda + kc + col8 * 8;
    async_copy_b128(loff, g);
  }
}

// Double-buffered pipelined GEMM core. Block = 64 threads (2 waves) sharing
// the LDS A strip; each wave computes a 16 x (16*NT) strip at its own n0.
template <int NT>
static __device__ __forceinline__ void gemm_pipelined(const _Float16* __restrict__ A, int lda,
                                                      const _Float16* __restrict__ W, int ldw,
                                                      int m0, int n0, int Kd,
                                                      _Float16* As, v8f acc[NT]) {
  stage_a_chunk(A, lda, m0, 0, As);
  wait_async0();
  __syncthreads();
  int buf = 0;
  for (int kc = 0; kc < Kd; kc += CHUNK) {
    if (kc + CHUNK < Kd)   // overlap next-chunk async copy with WMMA work
      stage_a_chunk(A, lda, m0, kc + CHUNK, As + ((buf ^ 1) * 16 * CHUNK));
    __builtin_prefetch(W + (size_t)(n0 + (threadIdx.x & 15)) * ldw + kc + CHUNK, 0, 0);
    const _Float16* Ab = As + buf * 16 * CHUNK;
#pragma unroll
    for (int kb = 0; kb < CHUNK; kb += 32) {
      v16h a = load_a_frag_lds(Ab, kb);
#pragma unroll
      for (int t = 0; t < NT; ++t) {
        v16h b = load_b_frag(W, ldw, n0 + 16 * t, kc + kb);
        acc[t] = __builtin_amdgcn_wmma_f32_16x16x32_f16(
            false, a, false, b, (short)0, acc[t], false, false);
      }
    }
    wait_async0();
    __syncthreads();
    buf ^= 1;
  }
}

// ---------------- elementwise / LN kernels ---------------------------------
__global__ void __launch_bounds__(256) k_f32_to_f16(const float* __restrict__ s,
                                                    _Float16* __restrict__ d, int n) {
  int i = blockIdx.x * 256 + threadIdx.x;
  if (i < n) d[i] = (_Float16)s[i];
}

// z[n, r, c] = base[r, c] + stream_bias[n, c]
__global__ void __launch_bounds__(256) k_make_z(const float* __restrict__ base,
                                                const float* __restrict__ sbias,
                                                float* __restrict__ z) {
  size_t i = (size_t)blockIdx.x * 256 + threadIdx.x;
  if (i >= (size_t)NBT * DMODEL) return;
  int c = (int)(i % DMODEL);
  size_t row = i / DMODEL;
  int n = (int)(row >> 12);            // row / BT
  size_t r = row & 4095;
  z[i] = base[r * DMODEL + c] + sbias[(size_t)n * DMODEL + c];
}

// rowwise LayerNorm over DMODEL, fp32 in -> f16 out
__global__ void __launch_bounds__(256) k_ln_rows(const float* __restrict__ in,
                                                 const float* __restrict__ g,
                                                 const float* __restrict__ b,
                                                 _Float16* __restrict__ out) {
  __shared__ float red[256];
  size_t row = blockIdx.x;
  const float* p = in + row * DMODEL;
  float s = 0.f;
  for (int c = threadIdx.x; c < DMODEL; c += 256) s += p[c];
  red[threadIdx.x] = s; __syncthreads();
  for (int o = 128; o > 0; o >>= 1) {
    if ((int)threadIdx.x < o) red[threadIdx.x] += red[threadIdx.x + o];
    __syncthreads();
  }
  float mean = red[0] * (1.0f / DMODEL);
  __syncthreads();
  float v = 0.f;
  for (int c = threadIdx.x; c < DMODEL; c += 256) { float d = p[c] - mean; v += d * d; }
  red[threadIdx.x] = v; __syncthreads();
  for (int o = 128; o > 0; o >>= 1) {
    if ((int)threadIdx.x < o) red[threadIdx.x] += red[threadIdx.x + o];
    __syncthreads();
  }
  float rstd = rsqrtf(red[0] * (1.0f / DMODEL) + LN_EPS);
  for (int c = threadIdx.x; c < DMODEL; c += 256)
    out[row * DMODEL + c] = (_Float16)((p[c] - mean) * rstd * g[c] + b[c]);
}

// ---------------- GEMM stage kernels (64 threads = 2 waves / block) --------
// base = x @ enc_W^T : (BT x IODIM) x (DMODEL x IODIM)^T
__global__ void __launch_bounds__(64) k_gemm_base(const _Float16* __restrict__ x16,
                                                  const _Float16* __restrict__ encW16,
                                                  float* __restrict__ base) {
  __shared__ _Float16 As[2 * 16 * CHUNK];
  v8f acc[4] = {};
  int m0 = blockIdx.x * 16;
  int n0 = blockIdx.y * 128 + ((threadIdx.x >> 5) << 6);
  gemm_pipelined<4>(x16, IODIM, encW16, IODIM, m0, n0, IODIM, As, acc);
  int lane = threadIdx.x & 31;
  int cb = lane & 15, rb = ((lane >> 4) & 1) * 8;
#pragma unroll
  for (int t = 0; t < 4; ++t)
#pragma unroll
    for (int r = 0; r < 8; ++r)
      base[(size_t)(m0 + rb + r) * DMODEL + n0 + 16 * t + cb] = acc[t][r];
}

// hmid = gelu(hln @ W1^T + b1), f16 out
__global__ void __launch_bounds__(64) k_gemm_mlp1(const _Float16* __restrict__ hln,
                                                  const _Float16* __restrict__ W1h,
                                                  const float* __restrict__ b1,
                                                  _Float16* __restrict__ hmid) {
  __shared__ _Float16 As[2 * 16 * CHUNK];
  v8f acc[4] = {};
  int m0 = blockIdx.x * 16;
  int n0 = blockIdx.y * 128 + ((threadIdx.x >> 5) << 6);
  gemm_pipelined<4>(hln, DMODEL, W1h, DMODEL, m0, n0, DMODEL, As, acc);
  int lane = threadIdx.x & 31;
  int cb = lane & 15, rb = ((lane >> 4) & 1) * 8;
#pragma unroll
  for (int t = 0; t < 4; ++t)
#pragma unroll
    for (int r = 0; r < 8; ++r) {
      int c = n0 + 16 * t + cb;
      float y = acc[t][r] + b1[c];
      float ge = 0.5f * y * (1.0f + erff(y * 0.70710678118654752f));  // exact gelu
      hmid[(size_t)(m0 + rb + r) * MIXHID + c] = (_Float16)ge;
    }
}

// zhead = z + hmid @ W2^T + b2, fp32 out
__global__ void __launch_bounds__(64) k_gemm_mlp2(const _Float16* __restrict__ hmid,
                                                  const _Float16* __restrict__ W2h,
                                                  const float* __restrict__ b2,
                                                  const float* __restrict__ z,
                                                  float* __restrict__ zhead) {
  __shared__ _Float16 As[2 * 16 * CHUNK];
  v8f acc[4] = {};
  int m0 = blockIdx.x * 16;
  int n0 = blockIdx.y * 128 + ((threadIdx.x >> 5) << 6);
  gemm_pipelined<4>(hmid, MIXHID, W2h, MIXHID, m0, n0, MIXHID, As, acc);
  int lane = threadIdx.x & 31;
  int cb = lane & 15, rb = ((lane >> 4) & 1) * 8;
#pragma unroll
  for (int t = 0; t < 4; ++t)
#pragma unroll
    for (int r = 0; r < 8; ++r) {
      int c = n0 + 16 * t + cb;
      size_t idx = (size_t)(m0 + rb + r) * DMODEL + c;
      zhead[idx] = z[idx] + acc[t][r] + b2[c];
    }
}

// QKV: per (which, n, kslot): (BT x DMODEL) x W[kslot*4+n] (DMODEL x DHEAD),
// scattered to the reference's scrambled (h, t2) head layout, f16 out.
__global__ void __launch_bounds__(64) k_gemm_qkv(const _Float16* __restrict__ zattn,
                                                 const _Float16* __restrict__ Wq16,
                                                 const _Float16* __restrict__ Wk16,
                                                 const _Float16* __restrict__ Wv16,
                                                 _Float16* __restrict__ q16,
                                                 _Float16* __restrict__ k16,
                                                 _Float16* __restrict__ v16o) {
  __shared__ _Float16 As[2 * 16 * CHUNK];
  int iz = blockIdx.z;
  int which = iz / 12;
  int rem = iz % 12;
  int n = rem / KSLOTS;
  int kslot = rem % KSLOTS;
  const _Float16* Wall = (which == 0) ? Wq16 : (which == 1) ? Wk16 : Wv16;
  const _Float16* W = Wall + (size_t)(kslot * N_STREAMS + n) * DMODEL * DHEAD;
  _Float16* dst = (which == 0) ? q16 : (which == 1) ? k16 : v16o;
  const _Float16* A = zattn + (size_t)n * BT * DMODEL;

  v8f acc[4] = {};
  int m0 = blockIdx.x * 16;
  int n0 = ((threadIdx.x >> 5) << 6);            // 2 waves cover all 128 cols
  gemm_pipelined<4>(A, DMODEL, W, DMODEL, m0, n0, DMODEL, As, acc);
  int lane = threadIdx.x & 31;
  int cb = lane & 15, rb = ((lane >> 4) & 1) * 8;
#pragma unroll
  for (int tt = 0; tt < 4; ++tt)
#pragma unroll
    for (int r = 0; r < 8; ++r) {
      int row = m0 + rb + r;                 // b*SEQ + t
      int bb = row >> 10;
      int tpos = row & 1023;
      int j = tpos * KSLOTS + kslot;          // reference reshape scramble
      int h = n * KSLOTS + (j >> 10);         // j / SEQ
      int t2 = j & 1023;                      // j % SEQ
      int dh = n0 + 16 * tt + cb;
      dst[((size_t)(bb * HEADS + h) * SEQ + t2) * DHEAD + dh] = (_Float16)acc[tt][r];
    }
}

// ---------------- attention: one wave per 16-row q-tile --------------------
__global__ void __launch_bounds__(32) k_attn(const _Float16* __restrict__ q16,
                                             const _Float16* __restrict__ k16,
                                             const _Float16* __restrict__ v16i,
                                             const int* __restrict__ is_causal,
                                             float* __restrict__ attn_out) {
  __shared__ float sm[16 * SEQ];              // 64 KB score strip (LDS: 320KB/WGP)
  int lane = threadIdx.x & 31;
  int qt0 = blockIdx.x * 16;
  int bh = blockIdx.y;                        // b*HEADS + h
  bool causal = (is_causal[0] != 0);
  const _Float16* Q = q16 + (size_t)bh * SEQ * DHEAD;
  const _Float16* Kp = k16 + (size_t)bh * SEQ * DHEAD;
  const _Float16* Vp = v16i + (size_t)bh * SEQ * DHEAD;
  const float scale = 0.088388347648318447f;  // 1/sqrt(128)
  int cb = lane & 15, rb = (lane >> 4) * 8;

  int ktend = causal ? (qt0 + 16) : SEQ;
  for (int kt0 = 0; kt0 < ktend; kt0 += 16) {
    v8f c = {};
#pragma unroll
    for (int kb = 0; kb < DHEAD; kb += 32) {
      v16h a = load_a_frag(Q, DHEAD, qt0, kb);
      v16h b = load_b_frag(Kp, DHEAD, kt0, kb);   // B[k][n] = K[kt0+n][k] == K^T
      c = __builtin_amdgcn_wmma_f32_16x16x32_f16(false, a, false, b, (short)0, c, false, false);
    }
#pragma unroll
    for (int r = 0; r < 8; ++r)
      sm[(rb + r) * SEQ + kt0 + cb] = c[r] * scale;
  }
  __syncthreads();

  int kmax = causal ? (((qt0 + 16) + 31) & ~31) : SEQ;
  if (kmax > SEQ) kmax = SEQ;
  if (lane < 16) {                            // fp32 softmax, one row per lane
    int m = lane;
    int qlim = causal ? (qt0 + m) : (SEQ - 1);
    float* row = sm + m * SEQ;
    float mx = -3.0e38f;
    for (int kk = 0; kk <= qlim; ++kk) mx = fmaxf(mx, row[kk]);
    float sum = 0.f;
    for (int kk = 0; kk <= qlim; ++kk) { float e = __expf(row[kk] - mx); row[kk] = e; sum += e; }
    float inv = 1.0f / sum;
    for (int kk = 0; kk <= qlim; ++kk) row[kk] *= inv;
    for (int kk = qlim + 1; kk < kmax; ++kk) row[kk] = 0.f;  // zero masked region
  }
  __syncthreads();

  v8f acc[8] = {};
  for (int kb = 0; kb < kmax; kb += 32) {
    int m = lane & 15;
    const float* pr = sm + m * SEQ + kb + ((lane >> 4) << 3);
    v16h a;
#pragma unroll
    for (int i = 0; i < 8; ++i) { a[i] = (_Float16)pr[i]; a[i + 8] = (_Float16)pr[i + 16]; }
    int ks = kb + ((lane >> 4) << 3);
#pragma unroll
    for (int t = 0; t < 8; ++t) {
      v16h b;
#pragma unroll
      for (int j = 0; j < 16; ++j)
        b[j] = Vp[(size_t)(ks + (j & 7) + ((j >> 3) << 4)) * DHEAD + t * 16 + cb];
      acc[t] = __builtin_amdgcn_wmma_f32_16x16x32_f16(false, a, false, b, (short)0, acc[t], false, false);
    }
  }
  float* O = attn_out + (size_t)bh * SEQ * DHEAD;
#pragma unroll
  for (int t = 0; t < 8; ++t)
#pragma unroll
    for (int r = 0; r < 8; ++r)
      O[(size_t)(qt0 + rb + r) * DHEAD + t * 16 + cb] = acc[t][r];
}

// ---------------- final merge: scatter active slots {0,1,3} ----------------
__global__ void __launch_bounds__(256) k_final(const float* __restrict__ zhead,
                                               const float* __restrict__ attn,
                                               float* __restrict__ out) {
  size_t i = (size_t)blockIdx.x * 256 + threadIdx.x;
  if (i >= (size_t)NBT * DMODEL) return;
  int d = (int)(i % DHEAD);
  size_t r = i / DHEAD;
  int s = (int)(r % S_SLOTS);
  size_t r2 = r / S_SLOTS;
  int t = (int)(r2 % SEQ);
  size_t r3 = r2 / SEQ;
  int b = (int)(r3 % BATCH);
  int n = (int)(r3 / BATCH);
  float v = zhead[i];
  int kidx = (s == 0) ? 0 : (s == 1) ? 1 : (s == 3) ? 2 : -1;  // ACTIVE0 = [0,1,3]
  if (kidx >= 0)
    v += attn[((size_t)(b * HEADS + n * KSLOTS + kidx) * SEQ + t) * DHEAD + d];
  out[i] = v;
}

// ---------------- host-side orchestration ----------------------------------
extern "C" void kernel_launch(void* const* d_in, const int* in_sizes, int n_in,
                              void* d_out, int out_size, void* d_ws, size_t ws_size,
                              hipStream_t stream) {
  (void)in_sizes; (void)n_in; (void)out_size; (void)ws_size;
  const float* x    = (const float*)d_in[0];
  const float* encW = (const float*)d_in[1];
  const float* sb   = (const float*)d_in[2];
  const float* gmx  = (const float*)d_in[3];
  const float* bmx  = (const float*)d_in[4];
  const float* W1   = (const float*)d_in[5];
  const float* b1   = (const float*)d_in[6];
  const float* W2   = (const float*)d_in[7];
  const float* b2   = (const float*)d_in[8];
  const float* ga   = (const float*)d_in[9];
  const float* ba   = (const float*)d_in[10];
  const float* Wq   = (const float*)d_in[11];
  const float* Wk   = (const float*)d_in[12];
  const float* Wv   = (const float*)d_in[13];
  const int* is_causal = (const int*)d_in[15];
  float* out = (float*)d_out;

  // workspace carve (256B aligned)
  char* w = (char*)d_ws;
  auto carve = [&](size_t bytes) -> void* {
    void* p = (void*)w;
    w += (bytes + 255) & ~(size_t)255;
    return p;
  };
  _Float16* x16    = (_Float16*)carve((size_t)BT * IODIM * 2);
  _Float16* encW16 = (_Float16*)carve((size_t)DMODEL * IODIM * 2);
  _Float16* W1h    = (_Float16*)carve((size_t)MIXHID * DMODEL * 2);
  _Float16* W2h    = (_Float16*)carve((size_t)DMODEL * MIXHID * 2);
  _Float16* Wq16   = (_Float16*)carve((size_t)HEADS * DMODEL * DHEAD * 2);
  _Float16* Wk16   = (_Float16*)carve((size_t)HEADS * DMODEL * DHEAD * 2);
  _Float16* Wv16   = (_Float16*)carve((size_t)HEADS * DMODEL * DHEAD * 2);
  float*    base   = (float*)   carve((size_t)BT * DMODEL * 4);
  float*    z      = (float*)   carve((size_t)NBT * DMODEL * 4);
  _Float16* hln    = (_Float16*)carve((size_t)NBT * DMODEL * 2);
  _Float16* hmid   = (_Float16*)carve((size_t)NBT * MIXHID * 2);
  float*    zhead  = (float*)   carve((size_t)NBT * DMODEL * 4);
  _Float16* zattn  = (_Float16*)carve((size_t)NBT * DMODEL * 2);
  _Float16* q16    = (_Float16*)carve((size_t)BATCH * HEADS * SEQ * DHEAD * 2);
  _Float16* k16    = (_Float16*)carve((size_t)BATCH * HEADS * SEQ * DHEAD * 2);
  _Float16* v16b   = (_Float16*)carve((size_t)BATCH * HEADS * SEQ * DHEAD * 2);
  float*    attnO  = (float*)   carve((size_t)BATCH * HEADS * SEQ * DHEAD * 4);

  auto cvt = [&](const float* s, _Float16* d, int n) {
    k_f32_to_f16<<<(n + 255) / 256, 256, 0, stream>>>(s, d, n);
  };
  cvt(x, x16, BT * IODIM);
  cvt(encW, encW16, DMODEL * IODIM);
  cvt(W1, W1h, MIXHID * DMODEL);
  cvt(W2, W2h, DMODEL * MIXHID);
  cvt(Wq, Wq16, HEADS * DMODEL * DHEAD);
  cvt(Wk, Wk16, HEADS * DMODEL * DHEAD);
  cvt(Wv, Wv16, HEADS * DMODEL * DHEAD);

  // 1. encoder GEMM (async-LDS pipelined, 2 waves/block)
  k_gemm_base<<<dim3(BT / 16, DMODEL / 128), 64, 0, stream>>>(x16, encW16, base);
  // 2. stream bias + LN(mix)
  int totD = NBT * DMODEL;
  k_make_z<<<(totD + 255) / 256, 256, 0, stream>>>(base, sb, z);
  k_ln_rows<<<NBT, 256, 0, stream>>>(z, gmx, bmx, hln);
  // 3. MLP
  k_gemm_mlp1<<<dim3(NBT / 16, MIXHID / 128), 64, 0, stream>>>(hln, W1h, b1, hmid);
  k_gemm_mlp2<<<dim3(NBT / 16, DMODEL / 128), 64, 0, stream>>>(hmid, W2h, b2, z, zhead);
  // 4. LN(attn) + QKV projections (36 GEMMs)
  k_ln_rows<<<NBT, 256, 0, stream>>>(zhead, ga, ba, zattn);
  k_gemm_qkv<<<dim3(BT / 16, 1, 36), 64, 0, stream>>>(
      zattn, Wq16, Wk16, Wv16, q16, k16, v16b);
  // 5. causal attention
  k_attn<<<dim3(SEQ / 16, BATCH * HEADS), 32, 0, stream>>>(q16, k16, v16b, is_causal, attnO);
  // 6. merge active slots into z_next
  k_final<<<(totD + 255) / 256, 256, 0, stream>>>(zhead, attnO, out);
}
